// Agent_63264868270168
// MI455X (gfx1250) — matrix operand order repaired and checked by reading, not compiled, not run on hardware
//
#include <hip/hip_runtime.h>

#define FN   128
#define KTOT 768
#define NKT  24
#define ROWS_PER_BLOCK 128
#define THREADS 256
#define XSS  28     // packed x floats per row (25 used + 3 zero pad)

typedef __attribute__((ext_vector_type(16))) __bf16 v16bf;
typedef __attribute__((ext_vector_type(8)))  float  v8f;

struct Params { const float* p[47]; };

union ABf { v16bf v; unsigned int u[8]; };

// Native RNE f32->bf16 via compiler casts (lowers to v_cvt_*bf16* on gfx1250,
// far cheaper than a manual add3/bfe RNE sequence).
__device__ __forceinline__ void split2(float f0, float f1,
                                       unsigned int& uh, unsigned int& ul) {
  union { __bf16 h[2]; unsigned int u; } H, L;
  H.h[0] = (__bf16)f0;
  H.h[1] = (__bf16)f1;
  L.h[0] = (__bf16)(f0 - (float)H.h[0]);
  L.h[1] = (__bf16)(f1 - (float)H.h[1]);
  uh = H.u; ul = L.u;
}
__device__ __forceinline__ void split_bf(float f, unsigned short& hi, unsigned short& lo) {
  union { __bf16 h; unsigned short s; } a, b;
  a.h = (__bf16)f;
  b.h = (__bf16)(f - (float)a.h);
  hi = a.s; lo = b.s;
}
__device__ __forceinline__ float sp_softplus(float v) {
  return (v > 20.f) ? v : log1pf(expf(v));
}

// 3 split-precision WMMAs: ah*bh + al*bh + ah*bl, f32 accumulate.
__device__ __forceinline__ void wmma3(const unsigned short* __restrict__ wsBh,
                                      const unsigned short* __restrict__ wsBl,
                                      int net, int kt, int lane,
                                      const ABf& ah, const ABf& al, v8f& acc) {
  const int boff = ((net*NKT + kt) << 9) + (lane << 4);
  ABf bh, bl;
  bh.v = *(const v16bf*)(wsBh + boff);
  bl.v = *(const v16bf*)(wsBl + boff);
  acc = __builtin_amdgcn_wmma_f32_16x16x32_bf16(false, ah.v, false, bh.v, (short)0, acc, false, false);
  acc = __builtin_amdgcn_wmma_f32_16x16x32_bf16(false, al.v, false, bh.v, (short)0, acc, false, false);
  acc = __builtin_amdgcn_wmma_f32_16x16x32_bf16(false, ah.v, false, bl.v, (short)0, acc, false, false);
}

// ---------------------------------------------------------------------------
// Fold kernel: build per-net folded B matrices [768 x 16] (bf16 hi/lo pair,
// stored in exact WMMA B-fragment lane layout) plus folded biases.
// Net order: 0=ar router(7 rows), 1=cr router(7), 2..4=actor experts (Wp@W4_e, 6),
// 5..7=critic experts (Wv@W4_e, 1). Unused rows are zero (harmless pad cols).
// ---------------------------------------------------------------------------
__global__ void __launch_bounds__(256)
moe_fold_kernel(Params P, unsigned short* wsBh, unsigned short* wsBl, float* wsBias) {
  const int net = blockIdx.x;
  const float* W4p = nullptr; const float* b4p = nullptr; int nrows = 0;
  const float* gptr[7] = {nullptr,nullptr,nullptr,nullptr,nullptr,nullptr,nullptr};
  float gbias[7] = {0,0,0,0,0,0,0};
  if (net < 2) {
    int b = net * 13;
    const float* Wtop = P.p[9+b];  const float* btop = P.p[10+b];
    const float* Wn   = P.p[11+b]; const float* bn   = P.p[12+b];
    const float* ps   = P.p[13+b];
    W4p = P.p[7+b]; b4p = P.p[8+b]; nrows = 7;
    for (int e = 0; e < 3; ++e) {
      gptr[e]   = Wtop + e*FN; gbias[e]   = btop[e];
      gptr[3+e] = Wn   + e*FN; gbias[3+e] = bn[e];
    }
    gptr[6] = ps; gbias[6] = 0.f;
  } else if (net < 5) {
    int e = net - 2;
    W4p = P.p[33] + (size_t)e*FN*KTOT; b4p = P.p[34] + e*FN; nrows = 6;
    for (int a = 0; a < 6; ++a) { gptr[a] = P.p[43] + a*FN; gbias[a] = P.p[44][a]; }
  } else {
    int e = net - 5;
    W4p = P.p[41] + (size_t)e*FN*KTOT; b4p = P.p[42] + e*FN; nrows = 1;
    gptr[0] = P.p[45]; gbias[0] = P.p[46][0];
  }
  for (int idx = threadIdx.x; idx < 16*KTOT; idx += blockDim.x) {
    int rN = idx / KTOT;
    int k  = idx - rN*KTOT;
    float f = 0.f;
    if (rN < nrows) {
      const float* g = gptr[rN];
      for (int n = 0; n < FN; ++n) f = fmaf(g[n], W4p[(size_t)n*KTOT + k], f);
    }
    // B fragment layout (32x16 bf16): lane = ((Klocal/8)%2)*16 + N, elem = Klocal%8 + (Klocal/16)*8
    int kt = k >> 5, Kl = k & 31;
    int lane = (((Kl >> 3) & 1) << 4) + rN;
    int e16  = (Kl & 7) + ((Kl >> 4) << 3);
    unsigned short hi, lo; split_bf(f, hi, lo);
    int off = ((net*NKT + kt) << 9) + (lane << 4) + e16;
    wsBh[off] = hi; wsBl[off] = lo;
  }
  if (threadIdx.x < 16) {
    int rN = threadIdx.x; float bsum = 0.f;
    if (rN < nrows) {
      bsum = gbias[rN];
      for (int n = 0; n < FN; ++n) bsum = fmaf(gptr[rN][n], b4p[n], bsum);
    }
    wsBias[net*16 + rN] = bsum;
  }
}

// ---------------------------------------------------------------------------
// Main fused kernel: 128 rows/block, 8 waves, each wave owns 16 rows.
// Per net: build merge (768 ReLU features) straight into WMMA A-fragment
// registers, multiply against folded 768x16 B via 24 x (3 split-precision)
// v_wmma_f32_16x16x32_bf16, dump 16x16 C tile to LDS, tiny per-row epilogues.
// Segment loop is runtime (6 iters) with 4 K-tiles unrolled inside to keep
// VGPR pressure under 256.
// ---------------------------------------------------------------------------
__global__ void __launch_bounds__(256)
moe_main_kernel(Params P, const unsigned short* __restrict__ wsBh,
                const unsigned short* __restrict__ wsBl,
                const float* __restrict__ wsBias, float* __restrict__ out, int Btot) {
  __shared__ float xs[ROWS_PER_BLOCK * XSS];
  __shared__ float swWfc[3*FN], swbfc[3*FN];
  __shared__ float swWcv[2*FN*8], swbcv[2*FN];
  __shared__ float swWc3[FN*8], swbc3[FN];
  __shared__ float scr[8][3][16][17];
  __shared__ int gidxA[ROWS_PER_BLOCK], gidxC[ROWS_PER_BLOCK];

  const int tid = threadIdx.x;
  const int lane = tid & 31;
  const int wv  = tid >> 5;
  const int hlf = lane >> 4;
  const int mm  = lane & 15;
  const int row0 = blockIdx.x * ROWS_PER_BLOCK;

  // Load packed x tile: [0]=x(r,0,7) [1]=x(r,1,7) [2]=x(r,5,7)
  // [3..10]=x(r,2,0..7) [11..18]=x(r,3,0..7) [19..24]=x(r,4,0..5) [25..27]=0
  if (tid < ROWS_PER_BLOCK) {
    int rr = row0 + tid; if (rr > Btot - 1) rr = Btot - 1;
    const float* xr = P.p[0] + (size_t)rr * 48;
    float* d = xs + tid * XSS;
    d[0] = xr[7]; d[1] = xr[15]; d[2] = xr[47];
    #pragma unroll
    for (int j = 0; j < 8; ++j) d[3+j]  = xr[16+j];
    #pragma unroll
    for (int j = 0; j < 8; ++j) d[11+j] = xr[24+j];
    #pragma unroll
    for (int j = 0; j < 6; ++j) d[19+j] = xr[32+j];
    d[25] = 0.f; d[26] = 0.f; d[27] = 0.f;
  }

  const float* xrow = xs + (wv*16 + mm) * XSS;

  for (int net = 0; net < 8; ++net) {
    __syncthreads();
    // stage this net's small weights into LDS
    {
      const float *Wfc, *bfc, *Wcv, *bcv, *Wc3, *bc3;
      if (net == 0)      { Wfc=P.p[1];  bfc=P.p[2];  Wcv=P.p[3];  bcv=P.p[4];  Wc3=P.p[5];  bc3=P.p[6];  }
      else if (net == 1) { Wfc=P.p[14]; bfc=P.p[15]; Wcv=P.p[16]; bcv=P.p[17]; Wc3=P.p[18]; bc3=P.p[19]; }
      else if (net < 5)  { int e=net-2; Wfc=P.p[27]+e*3*FN; bfc=P.p[28]+e*3*FN; Wcv=P.p[29]+e*2*FN*8;
                           bcv=P.p[30]+e*2*FN; Wc3=P.p[31]+e*FN*6; bc3=P.p[32]+e*FN; }
      else               { int e=net-5; Wfc=P.p[35]+e*3*FN; bfc=P.p[36]+e*3*FN; Wcv=P.p[37]+e*2*FN*8;
                           bcv=P.p[38]+e*2*FN; Wc3=P.p[39]+e*FN*6; bc3=P.p[40]+e*FN; }
      for (int i = tid; i < 3*FN; i += THREADS) { swWfc[i] = Wfc[i]; swbfc[i] = bfc[i]; }
      for (int i = tid; i < 2*FN*8; i += THREADS) swWcv[i] = Wcv[i];
      if (tid < 2*FN) swbcv[tid] = bcv[tid];
      if (tid < FN)   swbc3[tid] = bc3[tid];
      for (int i = tid; i < FN*8; i += THREADS) {
        int c = i >> 3, j = i & 7;
        swWc3[i] = (j < 6) ? Wc3[c*6 + j] : 0.f;   // pad Wc3 K=6 -> 8
      }
    }
    __syncthreads();

    v8f acc = {0.f,0.f,0.f,0.f,0.f,0.f,0.f,0.f};
    for (int seg = 0; seg < 6; ++seg) {
      if (seg == 0 || seg == 1 || seg == 5) {
        // scalar segments: s = relu(x_scalar * Wfc[fi] + bfc[fi])
        const int fi = (seg == 5) ? 2 : seg;
        const float t = xrow[fi];
        const float* wf  = swWfc + fi*FN;
        const float* bfv = swbfc + fi*FN;
        #pragma unroll
        for (int k4 = 0; k4 < 4; ++k4) {
          const int cbase = k4 << 5;
          ABf ah, al;
          #pragma unroll
          for (int v = 0; v < 8; ++v) {
            // A fragment 16x32 bf16: Klocal = (v&3)*2 + p + (lane/16)*8 + (v/4)*16
            int c0 = cbase + ((v & 3) << 1) + (hlf << 3) + ((v >> 2) << 4);
            float f0 = fmaxf(fmaf(t, wf[c0],   bfv[c0]),   0.f);
            float f1 = fmaxf(fmaf(t, wf[c0+1], bfv[c0+1]), 0.f);
            split2(f0, f1, ah.u[v], al.u[v]);
          }
          wmma3(wsBh, wsBl, net, seg*4 + k4, lane, ah, al, acc);
        }
      } else {
        // dot segments: s = relu(x8 . W[c] + b[c]) (Wc3 zero-padded to 8)
        const float *wd, *bd, *xv;
        if (seg == 2)      { wd = swWcv;        bd = swbcv;      xv = xrow + 3;  }
        else if (seg == 3) { wd = swWcv + FN*8; bd = swbcv + FN; xv = xrow + 11; }
        else               { wd = swWc3;        bd = swbc3;      xv = xrow + 19; }
        float x8[8];
        #pragma unroll
        for (int j = 0; j < 8; ++j) x8[j] = xv[j];
        #pragma unroll
        for (int k4 = 0; k4 < 4; ++k4) {
          const int cbase = k4 << 5;
          ABf ah, al;
          #pragma unroll
          for (int v = 0; v < 8; ++v) {
            int c0 = cbase + ((v & 3) << 1) + (hlf << 3) + ((v >> 2) << 4);
            float f0 = bd[c0], f1 = bd[c0+1];
            const float* w0 = wd + c0*8;
            #pragma unroll
            for (int j = 0; j < 8; ++j) { f0 = fmaf(x8[j], w0[j], f0); f1 = fmaf(x8[j], w0[8+j], f1); }
            f0 = fmaxf(f0, 0.f); f1 = fmaxf(f1, 0.f);
            split2(f0, f1, ah.u[v], al.u[v]);
          }
          wmma3(wsBh, wsBl, net, seg*4 + k4, lane, ah, al, acc);
        }
      }
    }

    // C layout: lane L, vgpr r -> (M = r + 8*(L/16), N = L%16)
    const int slot = (net < 2) ? net : ((net - 2) % 3);
    #pragma unroll
    for (int r = 0; r < 8; ++r) scr[wv][slot][r + hlf*8][mm] = acc[r];

    if (net == 1 || net == 4 || net == 7) {
      __syncthreads();
      if (tid < ROWS_PER_BLOCK) {
        const int m = tid, w8 = m >> 4, r16 = m & 15;
        const int row = row0 + m;
        if (net == 1) {
          // both routers done: top-1 gate via noisy logits
          #pragma unroll
          for (int rt = 0; rt < 2; ++rt) {
            const float* rb = wsBias + rt*16;
            float ph = scr[w8][rt][r16][6] + rb[6];
            float s  = sinf(ph);
            int bi = 0; float bv_ = -3.4e38f;
            #pragma unroll
            for (int e = 0; e < 3; ++e) {
              float l = scr[w8][rt][r16][e]   + rb[e];
              float n = scr[w8][rt][r16][3+e] + rb[3+e];
              float nv = l + s * sp_softplus(n);
              if (nv > bv_) { bv_ = nv; bi = e; }
            }
            if (rt == 0) gidxA[m] = bi; else gidxC[m] = bi;
          }
        } else if (net == 4) {
          // actor experts done: gated logits -> action/log_prob/entropy
          int e = gidxA[m];
          const float* pb = wsBias + (2 + e)*16;
          float lg[6]; float mx = -3.4e38f; int am = 0;
          #pragma unroll
          for (int a = 0; a < 6; ++a) {
            lg[a] = scr[w8][e][r16][a] + pb[a];
            if (lg[a] > mx) { mx = lg[a]; am = a; }
          }
          float se = 0.f;
          #pragma unroll
          for (int a = 0; a < 6; ++a) se += expf(lg[a] - mx);
          float lse = logf(se);
          float ent = 0.f;
          #pragma unroll
          for (int a = 0; a < 6; ++a) { float lp = lg[a] - mx - lse; ent -= expf(lp) * lp; }
          if (row < Btot) {
            out[row]          = (float)am;   // action (argmax)
            out[Btot + row]   = -lse;        // log_prob at argmax
            out[2*Btot + row] = ent;         // entropy
          }
        } else {
          // critic experts done: gated value
          int e = gidxC[m];
          float val = scr[w8][e][r16][0] + wsBias[(5 + e)*16];
          if (row < Btot) out[3*Btot + row] = val;
        }
      }
    }
  }
}

extern "C" void kernel_launch(void* const* d_in, const int* in_sizes, int n_in,
                              void* d_out, int out_size, void* d_ws, size_t ws_size,
                              hipStream_t stream) {
  (void)out_size; (void)ws_size;
  Params P;
  for (int i = 0; i < 47 && i < n_in; ++i) P.p[i] = (const float*)d_in[i];
  const int Btot = in_sizes[0] / 48;

  const int fragWords = 8 * NKT * 512;                 // ushort elements per plane
  unsigned short* wsBh = (unsigned short*)d_ws;
  unsigned short* wsBl = wsBh + fragWords;
  float* wsBias = (float*)(wsBl + fragWords);

  moe_fold_kernel<<<8, 256, 0, stream>>>(P, wsBh, wsBl, wsBias);
  const int nb = (Btot + ROWS_PER_BLOCK - 1) / ROWS_PER_BLOCK;
  moe_main_kernel<<<nb, 256, 0, stream>>>(P, wsBh, wsBl, wsBias, (float*)d_out, Btot);
}
